// QuantizedLinear_64355789964091
// MI455X (gfx1250) — compile-verified, matrix operand order checked by reading, not run on hardware
//
#include <hip/hip_runtime.h>

#define IN_F  4096
#define OUT_F 11008
#define NGRP  32

typedef __bf16 bf16;
typedef bf16 v16bf __attribute__((ext_vector_type(16)));
typedef bf16 v8bf  __attribute__((ext_vector_type(8)));
typedef float v8f  __attribute__((ext_vector_type(8)));

#define BM 128
#define BN 128
#define BK 64
#define KT_ITERS (IN_F / BK)            // 64
#define TSTRIDE 72                      // bf16 per tile row: 64 + 8 pad (16B align + bank rotate)
#define TILE_BYTES (128 * TSTRIDE * 2)  // 18432 B per tile buffer
#define C_STRIDE 132                    // f32 per C row: 128 + 4 pad

__device__ __forceinline__ v16bf combine16(v8bf a, v8bf b) {
  v16bf r;
  #pragma unroll
  for (int t = 0; t < 8; ++t) { r[t] = a[t]; r[8 + t] = b[t]; }
  return r;
}

__global__ __launch_bounds__(256)
void qlinear_wmma_bf16(const float* __restrict__ x,
                       const int*   __restrict__ qw,
                       const int*   __restrict__ qz,
                       const float* __restrict__ sc,
                       float*       __restrict__ out) {
  // 2x A tile + 2x B tile (double buffered); C staging overlays everything after the loop.
  __shared__ __align__(16) char smem[4 * TILE_BYTES];  // 73728 B

  const int tid   = threadIdx.x;
  const int lane  = tid & 31;
  const int wave  = tid >> 5;
  const int wm    = wave & 1;          // 0..1 -> 64 M rows
  const int wn    = wave >> 1;         // 0..3 -> 32 N cols
  const int lr    = lane & 15;
  const int lhalf = (lane >> 4) & 1;
  const int lo    = lhalf * 8;         // K sub-offset of bf16 fragment layout

  const int nBlock = blockIdx.x * BN;
  const int mBlock = blockIdx.y * BM;

  unsigned short* aBuf0 = (unsigned short*)(smem);
  unsigned short* aBuf1 = (unsigned short*)(smem + TILE_BYTES);
  unsigned short* bBuf0 = (unsigned short*)(smem + 2 * TILE_BYTES);
  unsigned short* bBuf1 = (unsigned short*)(smem + 3 * TILE_BYTES);

  // ---- per-thread staging assignments ----
  const int drow  = tid >> 1;          // B: 2 threads per weight row, 32 k each
  const int dhalf = tid & 1;
  const int o     = nBlock + drow;
  const float4* x4 = (const float4*)(x + (size_t)mBlock * IN_F);

  v8f acc[4][2] = {};

  float4 ra[8];   // in-flight A
  int4   rq[4];   // in-flight qweight
  float  rz = 0.f, rs = 0.f;

  auto issue_loads = [&](int kt) {
    #pragma unroll
    for (int jj = 0; jj < 4; ++jj) {
      int h = jj * 256 + tid;
      int row = h >> 3, c8 = h & 7;
      ra[2 * jj]     = x4[(size_t)row * (IN_F / 4) + kt * 16 + c8 * 2];
      ra[2 * jj + 1] = x4[(size_t)row * (IN_F / 4) + kt * 16 + c8 * 2 + 1];
    }
    const int4* qp = (const int4*)(qw + ((size_t)o * (IN_F / 2) + kt * (BK / 2) + dhalf * 16));
    #pragma unroll
    for (int jj = 0; jj < 4; ++jj) rq[jj] = qp[jj];
    if ((kt & 1) == 0) {               // group = kt/2: changes every other tile
      int g = kt >> 1;
      rz = (float)qz[o * NGRP + g];
      rs = sc[o * NGRP + g];
    }
  };

  auto commit = [&](unsigned short* aB, unsigned short* bB) {
    // A: cvt to bf16 (pairs fuse to v_cvt_pk_bf16_f32), 4x ds_store_b128
    #pragma unroll
    for (int jj = 0; jj < 4; ++jj) {
      int h = jj * 256 + tid;
      int row = h >> 3, c8 = h & 7;
      float4 f0 = ra[2 * jj], f1 = ra[2 * jj + 1];
      v8bf va;
      va[0] = (bf16)f0.x; va[1] = (bf16)f0.y; va[2] = (bf16)f0.z; va[3] = (bf16)f0.w;
      va[4] = (bf16)f1.x; va[5] = (bf16)f1.y; va[6] = (bf16)f1.z; va[7] = (bf16)f1.w;
      *(v8bf*)&aB[row * TSTRIDE + c8 * 8] = va;
    }
    // B: dequant nibbles -> bf16, 4x ds_store_b128
    float nz = -rz * rs;               // (w - z)*s = w*s + nz
    #pragma unroll
    for (int jj = 0; jj < 4; ++jj) {
      int q4[4] = {rq[jj].x, rq[jj].y, rq[jj].z, rq[jj].w};
      v8bf vb;
      #pragma unroll
      for (int t = 0; t < 4; ++t) {
        vb[2 * t]     = (bf16)fmaf((float)(q4[t] & 15), rs, nz);
        vb[2 * t + 1] = (bf16)fmaf((float)((q4[t] >> 4) & 15), rs, nz);
      }
      *(v8bf*)&bB[drow * TSTRIDE + dhalf * 32 + jj * 8] = vb;
    }
  };

  issue_loads(0);

  for (int kt = 0; kt < KT_ITERS; ++kt) {
    const int p = kt & 1;
    unsigned short* aB = p ? aBuf1 : aBuf0;
    unsigned short* bB = p ? bBuf1 : bBuf0;

    commit(aB, bB);                              // tile kt -> buffer p
    if (kt + 1 < KT_ITERS) issue_loads(kt + 1);  // next tile's global loads in flight
    __syncthreads();                             // tile kt visible to all waves

    // preload all fragments for both k-steps, then burst the 16 WMMAs
    v16bf aF[2][4], bF[2][2];
    #pragma unroll
    for (int ks = 0; ks < 2; ++ks) {
      const int kk = ks * 32;
      #pragma unroll
      for (int i = 0; i < 4; ++i) {
        const unsigned short* ab = &aB[(wm * 64 + i * 16 + lr) * TSTRIDE + kk + lo];
        aF[ks][i] = combine16(*(const v8bf*)(ab), *(const v8bf*)(ab + 16));
      }
      #pragma unroll
      for (int j = 0; j < 2; ++j) {
        const unsigned short* bb = &bB[(wn * 32 + j * 16 + lr) * TSTRIDE + kk + lo];
        bF[ks][j] = combine16(*(const v8bf*)(bb), *(const v8bf*)(bb + 16));
      }
    }
    #pragma unroll
    for (int ks = 0; ks < 2; ++ks)
      #pragma unroll
      for (int i = 0; i < 4; ++i)
        #pragma unroll
        for (int j = 0; j < 2; ++j)
          acc[i][j] = __builtin_amdgcn_wmma_f32_16x16x32_bf16(
              false, aF[ks][i], false, bF[ks][j], (short)0, acc[i][j], false, false);

    __syncthreads();                             // all reads of buffer p done before re-commit
  }

  // ---------------- epilogue: stage C in LDS, drain with async stores ----------------
  float* cBuf = (float*)smem;                    // 128 x C_STRIDE fp32, overlays tile buffers
  const unsigned cBase = (unsigned)(unsigned long long)(uintptr_t)cBuf;

  #pragma unroll
  for (int i = 0; i < 4; ++i) {
    #pragma unroll
    for (int j = 0; j < 2; ++j) {
      int m0 = wm * 64 + i * 16 + lhalf * 8;
      int n  = wn * 32 + j * 16 + lr;
      float* cp = cBuf + m0 * C_STRIDE + n;
      #pragma unroll
      for (int r = 0; r < 8; ++r) cp[r * C_STRIDE] = acc[i][j][r];
    }
  }
  __syncthreads();

  // 4096 float4 granules: thread handles 16; per-row 512B fully sequential in global
  #pragma unroll
  for (int jj = 0; jj < 16; ++jj) {
    int h = jj * 256 + tid;
    int row = h >> 5, c4 = h & 31;
    const float* gdst = out + (size_t)(mBlock + row) * OUT_F + nBlock + c4 * 4;
    unsigned lsrc = cBase + (unsigned)(row * (C_STRIDE * 4) + c4 * 16);
    asm volatile("global_store_async_from_lds_b128 %0, %1, off"
                 :: "v"(gdst), "v"(lsrc) : "memory");
  }
  asm volatile("s_wait_asynccnt 0" ::: "memory");
}

extern "C" void kernel_launch(void* const* d_in, const int* in_sizes, int n_in,
                              void* d_out, int out_size, void* d_ws, size_t ws_size,
                              hipStream_t stream) {
  const float* x  = (const float*)d_in[0];
  const int*   qw = (const int*)d_in[1];
  const int*   qz = (const int*)d_in[2];
  const float* sc = (const float*)d_in[3];
  float*       out = (float*)d_out;

  int M = in_sizes[0] / IN_F;            // 8192
  dim3 grid(OUT_F / BN, M / BM);         // 86 x 64
  qlinear_wmma_bf16<<<grid, dim3(256), 0, stream>>>(x, qw, qz, sc, out);
}
